// AffinityKernel_45732811767826
// MI455X (gfx1250) — matrix operand order repaired and checked by reading, not compiled, not run on hardware
//
#include <hip/hip_runtime.h>
#include <hip/hip_bf16.h>

// ---------------- problem constants ----------------
#define NPTS   16384
#define DIM    64      // IN_DIM
#define F2     128     // 2*IN_DIM
#define HID    128
#define OUTD   128
#define KNN    32

#define BIG    3.0e38f

typedef __attribute__((ext_vector_type(8)))  _Float16 v8h;
typedef __attribute__((ext_vector_type(16))) _Float16 v16h;
typedef __attribute__((ext_vector_type(8)))  float    v8f;

static __device__ __forceinline__ v16h cat16(v8h lo, v8h hi) {
  return __builtin_shufflevector(lo, hi, 0,1,2,3,4,5,6,7,8,9,10,11,12,13,14,15);
}

static __device__ __forceinline__ v8f wmma_f16(v16h a, v16h b, v8f c) {
  // D = A(16x32 f16) * B(32x16 f16) + C(16x16 f32)
  return __builtin_amdgcn_wmma_f32_16x16x32_f16(
      /*neg_a=*/false, a, /*neg_b=*/false, b,
      /*c_mod=*/(short)0, c, /*reuse_a=*/false, /*reuse_b=*/false);
}

static __device__ __forceinline__ void lds_fence() {
  asm volatile("s_wait_dscnt 0" ::: "memory");
}

// =====================================================================
// Kernel 1: sq[i] = |x_i|^2  and  xh = f16(x)
// =====================================================================
__global__ __launch_bounds__(256) void prep_kernel(const float* __restrict__ x,
                                                   float* __restrict__ sq,
                                                   _Float16* __restrict__ xh) {
  int row = blockIdx.x * blockDim.x + threadIdx.x;
  if (row >= NPTS) return;
  const float* xr = x + (size_t)row * DIM;
  _Float16* hr = xh + (size_t)row * DIM;
  float s = 0.f;
#pragma unroll
  for (int i = 0; i < DIM; i += 4) {
    float4 v = *(const float4*)(xr + i);
    s += v.x*v.x + v.y*v.y + v.z*v.z + v.w*v.w;
    hr[i+0] = (_Float16)v.x; hr[i+1] = (_Float16)v.y;
    hr[i+2] = (_Float16)v.z; hr[i+3] = (_Float16)v.w;
  }
  sq[row] = s;
}

// =====================================================================
// Kernel 2: fused Gram-tile (f16 WMMA) + per-row top-32 selection
//   8 waves/WG, 16 query rows per wave, 128 rows per WG, grid = 128 WGs
//   LDS carve (bytes):
//     sXT : 64 cols * 72 halves * 2      =  9216   (staged column tile)
//     sSq : 64 * 4                       =   256
//     sD2 : 8 waves * 256 * 4            =  8192   (d^2 tile per wave)
//     sVal: 256 threads * 32 * 4         = 32768
//     sIdx: 256 threads * 32 * 4         = 32768
//   total                                = 83200
// =====================================================================
#define XT_STRIDE 72   // halves; 144 B rows (16B aligned)
#define K2_LDS (9216 + 256 + 8192 + 32768 + 32768)

__global__ __launch_bounds__(256) void knn_kernel(const _Float16* __restrict__ xh,
                                                  const float* __restrict__ sq,
                                                  int* __restrict__ graph) {
  extern __shared__ char smem[];
  _Float16* sXT = (_Float16*)smem;                 // [64][XT_STRIDE]
  float*    sSq = (float*)(smem + 9216);           // [64]
  float*    sD2All = (float*)(smem + 9472);        // [8][16*16]
  float*    sVal = (float*)(smem + 17664);         // [256][KNN]
  int*      sIdx = (int*)(smem + 50432);           // [256][KNN]

  const int tid  = threadIdx.x;
  const int wave = tid >> 5;
  const int lane = tid & 31;
  const int hi   = lane >> 4;
  const int l15  = lane & 15;

  const int qbase = blockIdx.x * 128 + wave * 16;

  // ---- persistent A fragments for this wave's 16 query rows ----
  // A fragment (this code's contiguous k order): lane<16 -> row=l15, K = 16hi + 0..15
  v16h afrag[2];
  {
    const _Float16* rp = xh + (size_t)(qbase + l15) * DIM;
#pragma unroll
    for (int s = 0; s < 2; ++s) {
      const _Float16* p = rp + 32*s + 16*hi;
      v8h lo = *(const v8h*)p;
      v8h hi8 = *(const v8h*)(p + 8);
      afrag[s] = cat16(lo, hi8);
    }
  }

  // per-lane row norms in WMMA C layout: VGPR r -> row (r + 8*hi)
  float sqi[8];
#pragma unroll
  for (int r = 0; r < 8; ++r) sqi[r] = sq[qbase + r + 8*hi];

  // ---- selection state: lane (2r+half) owns row r, column half `half` ----
  const int selRow  = lane >> 1;
  const int selHalf = lane & 1;
  float* myVal = sVal + tid * KNN;
  int*   myIdx = sIdx + tid * KNN;
#pragma unroll
  for (int i = 0; i < KNN; ++i) { myVal[i] = BIG; myIdx[i] = 0; }
  float worst = BIG;
  int worstSlot = 0;
  float* dt = sD2All + wave * 256;

  for (int j0 = 0; j0 < NPTS; j0 += 64) {
    __syncthreads();
    {  // stage 64 columns of xh (+ their sq) into LDS
      int c  = tid >> 2;
      int fo = (tid & 3) * 16;
      const _Float16* src = xh + (size_t)(j0 + c) * DIM + fo;
      _Float16* dst = sXT + c * XT_STRIDE + fo;
      *(v8h*)dst       = *(const v8h*)src;
      *(v8h*)(dst + 8) = *(const v8h*)(src + 8);
      if (tid < 64) sSq[tid] = sq[j0 + tid];
    }
    __syncthreads();

#pragma unroll
    for (int sub = 0; sub < 4; ++sub) {
      const int jt = j0 + sub * 16;
      // B fragment: lane col = l15, K = 16hi + 0..15 (abs f = 32s + that)
      const _Float16* bp = sXT + (sub*16 + l15) * XT_STRIDE + 16*hi;
      v16h b0 = cat16(*(const v8h*)bp,        *(const v8h*)(bp + 8));
      v16h b1 = cat16(*(const v8h*)(bp + 32), *(const v8h*)(bp + 40));
      const float sqj = sSq[sub * 16 + l15];   // column norm for this lane
      v8f acc = {};
      acc = wmma_f16(afrag[0], b0, acc);
      acc = wmma_f16(afrag[1], b1, acc);
      // d^2 in-register, then spill tile: VGPR r, lane -> (row r+8hi, col l15)
#pragma unroll
      for (int r = 0; r < 8; ++r)
        dt[(r + 8*hi) * 16 + l15] = fmaf(-2.0f, acc[r], sqi[r] + sqj);
      lds_fence();
      // selection: 8 contiguous candidates -> two b128 loads
      const float4* dp = (const float4*)(dt + selRow * 16 + selHalf * 8);
      float4 d0 = dp[0];
      float4 d1 = dp[1];
      lds_fence();
      float dv[8] = {d0.x, d0.y, d0.z, d0.w, d1.x, d1.y, d1.z, d1.w};
      const int jb = jt + selHalf * 8;
#pragma unroll
      for (int c = 0; c < 8; ++c) {
        if (dv[c] < worst) {
          myVal[worstSlot] = dv[c];
          myIdx[worstSlot] = jb + c;
          // rescan for new worst (rare)
          worst = myVal[0]; worstSlot = 0;
#pragma unroll 4
          for (int i = 1; i < KNN; ++i) {
            float v = myVal[i];
            if (v > worst) { worst = v; worstSlot = i; }
          }
        }
      }
    }
  }

  // ---- merge the two 32-lists per row -> 32 smallest, write graph ----
  lds_fence();
  if (lane < 16) {
    const int row = qbase + lane;
    float* V = sVal + (wave * 32 + 2 * lane) * KNN;   // 64 contiguous entries
    int*   I = sIdx + (wave * 32 + 2 * lane) * KNN;
    int* g = graph + (size_t)row * KNN;
    for (int t = 0; t < KNN; ++t) {
      float best = V[0]; int bs = 0;
      for (int i = 1; i < 64; ++i) {
        float v = V[i];
        if (v < best) { best = v; bs = i; }
      }
      g[t] = I[bs];
      V[bs] = BIG;
    }
  }
}

// =====================================================================
// Kernel 3: gather -> feat(32x128 f16) -> clip(feat @ w.T) -> mean over k
//           -> out = [pooled, x] @ w2.T    (one point per wave)
//   LDS carve (bytes):
//     sW   : 128 * 136 halves * 2  = 34816
//     sF   : 8 * 32 * 136 * 2     = 69632
//     sPool: 8 * 128 * 4          =  4096
//   total                         = 108544
// =====================================================================
#define W_STRIDE 136   // halves; 272 B rows
#define F_STRIDE 136
#define K3_LDS (34816 + 69632 + 4096)

__global__ __launch_bounds__(256) void edgeconv_kernel(const float* __restrict__ x,
                                                       const _Float16* __restrict__ xh,
                                                       const float* __restrict__ w,
                                                       const float* __restrict__ w2,
                                                       const int* __restrict__ graph,
                                                       float* __restrict__ out) {
  extern __shared__ char smem[];
  _Float16* sW = (_Float16*)smem;                        // [128][W_STRIDE]
  _Float16* sFall = (_Float16*)(smem + 34816);           // [8][32][F_STRIDE]
  float*    sPoolAll = (float*)(smem + 34816 + 69632);   // [8][128]

  const int tid  = threadIdx.x;
  const int wave = tid >> 5;
  const int lane = tid & 31;
  const int hi   = lane >> 4;
  const int l15  = lane & 15;

  // stage w (f32 -> f16) once per WG
  for (int i = tid; i < HID * F2; i += 256) {
    int h = i >> 7, f = i & 127;
    sW[h * W_STRIDE + f] = (_Float16)w[h * F2 + f];
  }
  __syncthreads();

  _Float16* sF = sFall + wave * 32 * F_STRIDE;
  float* sPool = sPoolAll + wave * HID;

  const int stride = gridDim.x * 8;
  for (int n = blockIdx.x * 8 + wave; n < NPTS; n += stride) {
    // ---- build feature tile: row k = lane -> [x[nb] , x[n]] ----
    const int nb = graph[(size_t)n * KNN + lane];
    const _Float16* srcN = xh + (size_t)nb * DIM;
    const _Float16* srcS = xh + (size_t)n * DIM;
    _Float16* frow = sF + lane * F_STRIDE;
#pragma unroll
    for (int c = 0; c < 8; ++c) {
      *(v8h*)(frow + c * 8)      = *(const v8h*)(srcN + c * 8);
      *(v8h*)(frow + 64 + c * 8) = *(const v8h*)(srcS + c * 8);
    }
    lds_fence();

    // ---- hoist all A fragments (independent of output tile) ----
    v16h af[2][4];   // [m-tile][k-step]
#pragma unroll
    for (int mt = 0; mt < 2; ++mt) {
#pragma unroll
      for (int s = 0; s < 4; ++s) {
        const _Float16* ap = sF + (l15 + 16*mt) * F_STRIDE + 32*s + 16*hi;
        af[mt][s] = cat16(*(const v8h*)ap, *(const v8h*)(ap + 8));
      }
    }

    // ---- H = clip(F * W^T), pooled = mean over 32 rows ----
#pragma unroll
    for (int nt = 0; nt < 8; ++nt) {
      v8f acc0 = {}, acc1 = {};
#pragma unroll
      for (int s = 0; s < 4; ++s) {
        const _Float16* bp = sW + (nt * 16 + l15) * W_STRIDE + 32 * s + 16 * hi;
        v16h b = cat16(*(const v8h*)bp, *(const v8h*)(bp + 8));
        acc0 = wmma_f16(af[0][s], b, acc0);
        acc1 = wmma_f16(af[1][s], b, acc1);
      }
      float ssum = 0.f;
#pragma unroll
      for (int r = 0; r < 8; ++r) {
        ssum += fminf(fmaxf(acc0[r], -1.f), 1.f);
        ssum += fminf(fmaxf(acc1[r], -1.f), 1.f);
      }
      ssum += __shfl_xor(ssum, 16, 32);   // combine row halves of same column
      if (hi == 0) sPool[nt * 16 + l15] = ssum * (1.f / 32.f);
    }
    lds_fence();

    // ---- out[n] = [pooled, x[n]] @ w2.T  (lane -> 4 outputs) ----
    const float* xr = x + (size_t)n * DIM;
#pragma unroll
    for (int q = 0; q < 4; ++q) {
      const int o = lane + q * 32;
      const float* w2r = w2 + (size_t)o * (HID + DIM);
      float acc = 0.f;
      for (int h = 0; h < HID; ++h) acc = fmaf(sPool[h], w2r[h], acc);
      for (int f = 0; f < DIM; ++f) acc = fmaf(xr[f], w2r[HID + f], acc);
      out[(size_t)n * OUTD + o] = acc;
    }
    lds_fence();
  }
}

// =====================================================================
extern "C" void kernel_launch(void* const* d_in, const int* in_sizes, int n_in,
                              void* d_out, int out_size, void* d_ws, size_t ws_size,
                              hipStream_t stream) {
  const float* x  = (const float*)d_in[0];   // (N, 64)
  const float* w  = (const float*)d_in[1];   // (128, 128)
  const float* w2 = (const float*)d_in[2];   // (128, 192)
  (void)in_sizes; (void)n_in; (void)out_size; (void)ws_size;
  float* out = (float*)d_out;                // (N, 128)

  // workspace carve
  char* ws = (char*)d_ws;
  float*    sq    = (float*)ws;                              // N f32      (64 KB)
  _Float16* xh    = (_Float16*)(ws + 65536);                 // N*64 f16   (2 MB)
  int*      graph = (int*)(ws + 65536 + (size_t)NPTS*DIM*2); // N*32 int   (2 MB)

  prep_kernel<<<dim3(NPTS / 256), dim3(256), 0, stream>>>(x, sq, xh);
  knn_kernel<<<dim3(NPTS / 128), dim3(256), K2_LDS, stream>>>(xh, sq, graph);
  edgeconv_kernel<<<dim3(512), dim3(256), K3_LDS, stream>>>(x, xh, w, w2, graph, out);
}
